// Mask2FormerForgeryModel_88201448390735
// MI455X (gfx1250) — compile-verified, matrix operand order checked by reading, not run on hardware
//
#include <hip/hip_runtime.h>
#include <hip/hip_bf16.h>

// ---------------------------------------------------------------------------
// Multi-scale deformable attention (MSDeformAttn) for gfx1250 (MI455X).
// B=2, S=21504, C=256, NH=8, L=3, P=4, D=32. wave32, WMMA f16->f32.
// ---------------------------------------------------------------------------

#define BDIM   2
#define SDIM   21504
#define CDIM   256
#define NHDIM  8
#define LDIM   3
#define PDIM   4
#define DDIM   32
#define ROWS   (BDIM * SDIM)      // 43008

typedef __attribute__((ext_vector_type(16))) _Float16 v16h;
typedef __attribute__((ext_vector_type(8)))  _Float16 h8;
typedef __attribute__((ext_vector_type(8)))  float    v8f;

// ---------------------------------------------------------------------------
// Generic GEMM: Y[rows x N] = X[rows x 256] * W[256 x N] + bias[N]
// fp32 in/out, f16 WMMA compute (fp32 accumulate inside the matrix pipe).
// Block: 256 threads = 8 waves; block tile 128(M) x 32(N).
// Wave w -> 16x32 output (M = 16*w): TWO accumulators sharing one A
// fragment per K-step => 2 WMMAs per 6 LDS fragment loads.
// ---------------------------------------------------------------------------
#define LDA 40   // padded f16 row stride (80B, 16B aligned, spreads banks)
#define LDB 40

__global__ __launch_bounds__(256)
void wmma_gemm_f16(const float* __restrict__ X, const float* __restrict__ W,
                   const float* __restrict__ bias, float* __restrict__ Y,
                   int N)
{
    __shared__ _Float16 Xs[128 * LDA];  // A tile [128][32] row-major
    __shared__ _Float16 Wt[32 * LDB];   // B tile transposed: [n][k]

    const int tid  = threadIdx.x;
    const int wave = tid >> 5;
    const int lane = tid & 31;
    const int row0 = blockIdx.x * 128;
    const int n0   = blockIdx.y * 32;
    const int mT   = wave * 16;         // wave M offset within block tile

    const int half_sel = lane >> 4;     // 0: K 0-7/16-23, 1: K 8-15/24-31
    const int kb       = half_sel * 8;
    const int mfrag    = mT + (lane & 15);
    const int nfrag    = lane & 15;

    v8f acc0 = {};   // columns n0 + 0..15
    v8f acc1 = {};   // columns n0 + 16..31

    // staging coordinates (constant across K loop)
    const int rX  = tid >> 1;            // 0..127
    const int cX  = (tid & 1) << 4;      // 0 or 16
    const int kW  = tid >> 3;            // 0..31
    const int nW  = (tid & 7) << 2;      // 0,4,...,28

    for (int k0 = 0; k0 < CDIM; k0 += 32) {
        __syncthreads();
        // --- stage A: 128x32 fp32 -> f16, row-major (16 elems/thread) ---
        {
            const float* src = X + (size_t)(row0 + rX) * CDIM + k0 + cX;
            #pragma unroll
            for (int i = 0; i < 16; ++i)
                Xs[rX * LDA + cX + i] = (_Float16)src[i];
        }
        // --- stage B: 32x32 fp32 (row-major W[k][n]) -> f16 transposed ---
        {
            const float* src = W + (size_t)(k0 + kW) * N + n0 + nW;
            #pragma unroll
            for (int i = 0; i < 4; ++i)
                Wt[(nW + i) * LDB + kW] = (_Float16)src[i];
        }
        __syncthreads();

        // prefetch next K tile of A into cache (global_prefetch_b8)
        if (k0 + 32 < CDIM)
            __builtin_prefetch(X + (size_t)(row0 + rX) * CDIM + k0 + 32 + cX, 0, 1);

        // --- fragments per the CDNA5 16-bit A/B VGPR layouts ---
        h8 alo  = *(const h8*)&Xs[mfrag * LDA + kb];        // K = kb..kb+7
        h8 ahi  = *(const h8*)&Xs[mfrag * LDA + kb + 16];   // K = kb+16..kb+23
        h8 b0lo = *(const h8*)&Wt[nfrag * LDB + kb];
        h8 b0hi = *(const h8*)&Wt[nfrag * LDB + kb + 16];
        h8 b1lo = *(const h8*)&Wt[(nfrag + 16) * LDB + kb];
        h8 b1hi = *(const h8*)&Wt[(nfrag + 16) * LDB + kb + 16];
        v16h a  = __builtin_shufflevector(alo,  ahi,  0,1,2,3,4,5,6,7,8,9,10,11,12,13,14,15);
        v16h b0 = __builtin_shufflevector(b0lo, b0hi, 0,1,2,3,4,5,6,7,8,9,10,11,12,13,14,15);
        v16h b1 = __builtin_shufflevector(b1lo, b1hi, 0,1,2,3,4,5,6,7,8,9,10,11,12,13,14,15);

        acc0 = __builtin_amdgcn_wmma_f32_16x16x32_f16(
            false, a, false, b0, (short)0, acc0, false, false);
        acc1 = __builtin_amdgcn_wmma_f32_16x16x32_f16(
            false, a, false, b1, (short)0, acc1, false, false);
    }

    // --- store: C/D layout: VGPR r -> M = r + 8*half_sel, N = lane&15 ---
    const int ncol0 = n0 + nfrag;
    const int ncol1 = ncol0 + 16;
    const float bn0 = bias[ncol0];
    const float bn1 = bias[ncol1];
    const int mrow  = row0 + mT + 8 * half_sel;
    #pragma unroll
    for (int r = 0; r < 8; ++r) {
        Y[(size_t)(mrow + r) * N + ncol0] = acc0[r] + bn0;
        Y[(size_t)(mrow + r) * N + ncol1] = acc1[r] + bn1;
    }
}

// ---------------------------------------------------------------------------
// Deformable sampling + softmax(P) + head-weighted sum.
// One wave per (b, s, head). Lane = channel d (D = 32 = wave32).
// Lanes 0..11 own one (level, point) each: location + softmax over P=4
// via wave32 __shfl_xor. Then all lanes gather 4 bilinear corners per
// sample (coalesced 128B reads, v fully L2-resident).
// ---------------------------------------------------------------------------
__global__ __launch_bounds__(256)
void msda_sample(const float* __restrict__ v,    // [ROWS][256] col = h*32+d
                 const float* __restrict__ off,  // [ROWS][192]
                 const float* __restrict__ awl,  // [ROWS][96] logits
                 const float* __restrict__ refp, // [B,S,L,2]
                 float* __restrict__ out)        // [ROWS][256]
{
    constexpr int LH[3]  = {128, 64, 32};
    constexpr int LW[3]  = {128, 64, 32};
    constexpr int LST[3] = {0, 16384, 20480};

    const int lane = threadIdx.x & 31;
    const int h    = (threadIdx.x >> 5) & 7;  // 8 waves per block = 8 heads
    const int row  = blockIdx.x;              // b*S + s
    const int b    = row / SDIM;

    // ---- phase 1: lanes 0..11 compute sample (x, y, weight) ----
    float sx = 0.f, sy = 0.f, sw = 0.f;
    if (lane < LDIM * PDIM) {
        const int l = lane >> 2, p = lane & 3;
        const float rx = refp[((size_t)row * LDIM + l) * 2 + 0];
        const float ry = refp[((size_t)row * LDIM + l) * 2 + 1];
        const float ox = off[(size_t)row * 192 + h * 24 + l * 8 + p * 2 + 0];
        const float oy = off[(size_t)row * 192 + h * 24 + l * 8 + p * 2 + 1];
        const float lg = awl[(size_t)row * 96 + h * 12 + l * 4 + p];

        // softmax over the 4 lanes of this (h, l) group (group-aligned xor)
        float mx = fmaxf(lg, __shfl_xor(lg, 1, 32));
        mx = fmaxf(mx, __shfl_xor(mx, 2, 32));
        float e = __expf(lg - mx);
        float sum = e + __shfl_xor(e, 1, 32);
        sum = sum + __shfl_xor(sum, 2, 32);
        sw = e / sum;

        // sample position in pixel coords (grid_sample align_corners=False):
        // x = (ref_x + off_x / W) * W - 0.5 = ref_x*W + off_x - 0.5
        sx = rx * (float)LW[l] + ox - 0.5f;
        sy = ry * (float)LH[l] + oy - 0.5f;
    }

    // ---- phase 2: all 32 lanes (= 32 channels) accumulate 12 samples ----
    float accum = 0.f;
    #pragma unroll
    for (int j = 0; j < LDIM * PDIM; ++j) {
        const int l  = j >> 2;
        const int Wl = LW[l], Hl = LH[l];
        const float x = __shfl(sx, j, 32);
        const float y = __shfl(sy, j, 32);
        const float w = __shfl(sw, j, 32);

        const float x0f = floorf(x), y0f = floorf(y);
        const int   x0  = (int)x0f,  y0  = (int)y0f;
        const float fx  = x - x0f,   fy  = y - y0f;

        const float* vb = v + ((size_t)(b * SDIM + LST[l])) * CDIM + h * DDIM + lane;

        float samp = 0.f;
        #pragma unroll
        for (int cy = 0; cy < 2; ++cy) {
            #pragma unroll
            for (int cx = 0; cx < 2; ++cx) {
                const int xi = x0 + cx, yi = y0 + cy;
                const bool valid = (xi >= 0) && (xi < Wl) && (yi >= 0) && (yi < Hl);
                const int xc = min(max(xi, 0), Wl - 1);
                const int yc = min(max(yi, 0), Hl - 1);
                const float cw = (cx ? fx : 1.f - fx) * (cy ? fy : 1.f - fy);
                const float val = vb[(size_t)(yc * Wl + xc) * CDIM];  // coalesced 128B
                samp += valid ? cw * val : 0.f;
            }
        }
        accum += w * samp;
    }

    out[(size_t)row * CDIM + h * DDIM + lane] = accum;
}

// ---------------------------------------------------------------------------
// Launch. Inputs (setup_inputs order):
//  0 query [B,S,C]  1 value [B,S,C]  2 reference_points [B,S,L,2]
//  3 Wv [C,C] 4 bv [C] 5 Ws [C,192] 6 bs [192] 7 Wa [C,96] 8 ba [96]
//  9 Wo [C,C] 10 bo [C]
// ---------------------------------------------------------------------------
extern "C" void kernel_launch(void* const* d_in, const int* in_sizes, int n_in,
                              void* d_out, int out_size, void* d_ws, size_t ws_size,
                              hipStream_t stream) {
    const float* query = (const float*)d_in[0];
    const float* value = (const float*)d_in[1];
    const float* refp  = (const float*)d_in[2];
    const float* Wv    = (const float*)d_in[3];
    const float* bv    = (const float*)d_in[4];
    const float* Ws    = (const float*)d_in[5];
    const float* bs    = (const float*)d_in[6];
    const float* Wa    = (const float*)d_in[7];
    const float* ba    = (const float*)d_in[8];
    const float* Wo    = (const float*)d_in[9];
    const float* bo    = (const float*)d_in[10];
    float* outp = (float*)d_out;

    // workspace layout (floats): v | off | aw | att
    float* wsf = (float*)d_ws;
    float* v   = wsf;
    float* off = v   + (size_t)ROWS * 256;
    float* aw  = off + (size_t)ROWS * 192;
    float* att = aw  + (size_t)ROWS * 96;

    const dim3 blk(256);

    // value projection: [ROWS,256] @ [256,256]
    wmma_gemm_f16<<<dim3(ROWS / 128, 256 / 32), blk, 0, stream>>>(value, Wv, bv, v, 256);
    // sampling offsets: [ROWS,256] @ [256,192]
    wmma_gemm_f16<<<dim3(ROWS / 128, 192 / 32), blk, 0, stream>>>(query, Ws, bs, off, 192);
    // attention logits: [ROWS,256] @ [256,96]
    wmma_gemm_f16<<<dim3(ROWS / 128, 96 / 32), blk, 0, stream>>>(query, Wa, ba, aw, 96);
    // deformable sampling + softmax + head mix: one block per (b,s), 8 waves = 8 heads
    msda_sample<<<dim3(ROWS), blk, 0, stream>>>(v, off, aw, refp, att);
    // output projection: [ROWS,256] @ [256,256] -> d_out
    wmma_gemm_f16<<<dim3(ROWS / 128, 256 / 32), blk, 0, stream>>>(att, Wo, bo, outp, 256);
}
